// Transformer_79018808312501
// MI455X (gfx1250) — matrix operand order repaired and checked by reading, not compiled, not run on hardware
//
#include <hip/hip_runtime.h>
#include <math.h>

typedef _Float16 v16h __attribute__((ext_vector_type(16)));
typedef float    v8f  __attribute__((ext_vector_type(8)));

#define T_TOK 2048
#define DDIM  2048
#define IDIM  4096
#define NEXP  8
#define RMAX  4608   // 4096 pairs + 8*63 alignment padding, rounded to 64

// ---- workspace layout (bytes) ----
#define WS_COUNTS   0                         // 8 ints
#define WS_CURSORS  32                        // 8 ints (contiguous with counts)
#define WS_AOFF     64                        // 9 ints
#define WS_TOKE     128                       // T*2 ints
#define WS_TOKW     (128 + 16384)             // T*2 floats
#define WS_ROWTOK   (WS_TOKW + 16384)         // RMAX ints
#define WS_ROWW     (WS_ROWTOK + 18432)       // RMAX floats
#define WS_ROWSLOT  (WS_ROWW + 18432)         // RMAX ints
#define WS_H        88320                     // RMAX*IDIM f16  (37,748,736 B)
#define WS_PARTIAL  37837056                  // T*2*DDIM f32  (33,554,432 B)

// -------------------- WMMA fragment loaders (LDS, f16) --------------------
// A tile in LDS: [64 rows(M)][32 cols(K)] f16. A frag 16x32:
//   lanes 0-15: M=lane, K=0..7 (v0..3) & K=16..23 (v4..7)
//   lanes16-31: M=lane-16, K=8..15 & K=24..31
static __device__ inline v16h load_frag_a(const _Float16* s, int m_off, int lane) {
  const _Float16* p = s + (m_off + (lane & 15)) * 32 + ((lane >> 4) & 1) * 8;
  v16h r;
  ((float4*)&r)[0] = *(const float4*)(p);
  ((float4*)&r)[1] = *(const float4*)(p + 16);
  return r;
}
// B tile in LDS: [64 rows(N)][32 cols(K)] f16. B frag 32x16:
//   lanes 0-15: N=lane, K=0..7 & K=16..23 ; lanes16-31: N=lane-16, K=8..15 & K=24..31
static __device__ inline v16h load_frag_b(const _Float16* s, int n_off, int lane) {
  const _Float16* p = s + (n_off + (lane & 15)) * 32 + ((lane >> 4) & 1) * 8;
  v16h r;
  ((float4*)&r)[0] = *(const float4*)(p);
  ((float4*)&r)[1] = *(const float4*)(p + 16);
  return r;
}

// -------------------- routing kernels --------------------
__global__ void moe_init(int* hdr16, int* row_token) {
  int i = blockIdx.x * blockDim.x + threadIdx.x;
  if (i < 16) hdr16[i] = 0;            // counts + cursors
  if (i < RMAX) row_token[i] = -1;
}

__global__ void moe_gate(const float* __restrict__ x, const float* __restrict__ gw,
                         int* counts, int* tok_e, float* tok_w) {
  int lane = threadIdx.x & 31;
  int wave = threadIdx.x >> 5;
  int t = blockIdx.x * 8 + wave;
  if (t >= T_TOK) return;
  const float* xr = x + (size_t)t * DDIM;
  float acc[NEXP];
#pragma unroll
  for (int e = 0; e < NEXP; ++e) acc[e] = 0.f;
  for (int d = lane; d < DDIM; d += 32) {
    float xv = xr[d];
#pragma unroll
    for (int e = 0; e < NEXP; ++e) acc[e] += xv * gw[e * DDIM + d];
  }
#pragma unroll
  for (int e = 0; e < NEXP; ++e) {
#pragma unroll
    for (int off = 16; off > 0; off >>= 1)
      acc[e] += __shfl_xor(acc[e], off, 32);
  }
  if (lane == 0) {
    int i0 = 0;
#pragma unroll
    for (int e = 1; e < NEXP; ++e) if (acc[e] > acc[i0]) i0 = e;
    int i1 = (i0 == 0) ? 1 : 0;
#pragma unroll
    for (int e = 0; e < NEXP; ++e) if (e != i0 && acc[e] > acc[i1]) i1 = e;
    // softmax + top-2 renorm collapses to pairwise softmax of the two scores
    float w0 = 1.f / (1.f + __expf(acc[i1] - acc[i0]));
    tok_e[t * 2 + 0] = i0; tok_e[t * 2 + 1] = i1;
    tok_w[t * 2 + 0] = w0; tok_w[t * 2 + 1] = 1.f - w0;
    atomicAdd(&counts[i0], 1);
    atomicAdd(&counts[i1], 1);
  }
}

__global__ void moe_scan(const int* counts, int* aoff) {
  if (threadIdx.x == 0 && blockIdx.x == 0) {
    int acc = 0;
    aoff[0] = 0;
#pragma unroll
    for (int e = 0; e < NEXP; ++e) { acc += ((counts[e] + 63) >> 6) << 6; aoff[e + 1] = acc; }
  }
}

__global__ void moe_scatter(const int* __restrict__ tok_e, const float* __restrict__ tok_w,
                            const int* __restrict__ aoff, int* cursors,
                            int* row_token, float* row_w, int* row_slot) {
  int t = blockIdx.x * blockDim.x + threadIdx.x;
  if (t >= T_TOK) return;
#pragma unroll
  for (int a = 0; a < 2; ++a) {
    int e = tok_e[t * 2 + a];
    int pos = aoff[e] + atomicAdd(&cursors[e], 1);
    row_token[pos] = t;
    row_w[pos] = tok_w[t * 2 + a];
    row_slot[pos] = a;
  }
}

// -------------------- GEMM 1: H = silu(X*W1^T) * (X*W3^T) --------------------
__global__ __launch_bounds__(128)
void moe_ffn1(const float* __restrict__ x,
              const float* __restrict__ w1, const float* __restrict__ w3,
              const int* __restrict__ aoff, const int* __restrict__ counts,
              const int* __restrict__ row_token, _Float16* __restrict__ H) {
  __shared__ __align__(16) _Float16 sA[2][64 * 32];
  __shared__ __align__(16) _Float16 sB1[2][64 * 32];
  __shared__ __align__(16) _Float16 sB3[2][64 * 32];
  __shared__ int sTok[64];

  const int m0 = blockIdx.y * 64;   // padded-row tile base
  const int n0 = blockIdx.x * 64;   // I-channel tile base
  if (m0 >= aoff[8]) return;
  int e = 0;
#pragma unroll
  for (int k = 1; k < NEXP; ++k) if (m0 >= aoff[k]) e = k;
  if (m0 - aoff[e] >= counts[e]) return;   // fully-padded tile

  const int tid = threadIdx.x;
  const int lane = tid & 31;
  const int wave = tid >> 5;
  const int m_sub = (wave & 1) * 32;
  const int n_sub = (wave >> 1) * 32;
  const float* w1e = w1 + (size_t)e * IDIM * DDIM;
  const float* w3e = w3 + (size_t)e * IDIM * DDIM;

  if (tid < 64) { int tk = row_token[m0 + tid]; sTok[tid] = (tk < 0) ? 0 : tk; }
  __syncthreads();

  // static per-thread load decomposition: 512 float4 eighth-rows / 128 threads
  int rowt[4], colt[4];
#pragma unroll
  for (int it = 0; it < 4; ++it) {
    int task = tid + it * 128;
    rowt[it] = task >> 3;
    colt[it] = (task & 7) << 2;
  }

  float4 pa[4], p1[4], p3[4];
  auto prefetch = [&](int k0) {
#pragma unroll
    for (int it = 0; it < 4; ++it) {
      pa[it] = *(const float4*)(x   + (size_t)sTok[rowt[it]] * DDIM + k0 + colt[it]);
      p1[it] = *(const float4*)(w1e + (size_t)(n0 + rowt[it]) * DDIM + k0 + colt[it]);
      p3[it] = *(const float4*)(w3e + (size_t)(n0 + rowt[it]) * DDIM + k0 + colt[it]);
    }
  };
  auto stage = [&](int b) {
#pragma unroll
    for (int it = 0; it < 4; ++it) {
      int o = rowt[it] * 32 + colt[it];
      sA[b][o + 0]  = (_Float16)pa[it].x; sA[b][o + 1]  = (_Float16)pa[it].y;
      sA[b][o + 2]  = (_Float16)pa[it].z; sA[b][o + 3]  = (_Float16)pa[it].w;
      sB1[b][o + 0] = (_Float16)p1[it].x; sB1[b][o + 1] = (_Float16)p1[it].y;
      sB1[b][o + 2] = (_Float16)p1[it].z; sB1[b][o + 3] = (_Float16)p1[it].w;
      sB3[b][o + 0] = (_Float16)p3[it].x; sB3[b][o + 1] = (_Float16)p3[it].y;
      sB3[b][o + 2] = (_Float16)p3[it].z; sB3[b][o + 3] = (_Float16)p3[it].w;
    }
  };

  const v8f vzero = {0.f, 0.f, 0.f, 0.f, 0.f, 0.f, 0.f, 0.f};
  v8f acc1[2][2], acc3[2][2];
#pragma unroll
  for (int mi = 0; mi < 2; ++mi)
#pragma unroll
    for (int ni = 0; ni < 2; ++ni) { acc1[mi][ni] = vzero; acc3[mi][ni] = vzero; }

  prefetch(0);
  stage(0);
  int cur = 0;

  for (int k0 = 0; k0 < DDIM; k0 += 32) {
    __syncthreads();                       // publish buf[cur]
    const bool more = (k0 + 32) < DDIM;
    if (more) prefetch(k0 + 32);           // overlap next global fetch with WMMA

    v16h afr[2], b1fr[2], b3fr[2];
#pragma unroll
    for (int mi = 0; mi < 2; ++mi) afr[mi] = load_frag_a(sA[cur], m_sub + 16 * mi, lane);
#pragma unroll
    for (int ni = 0; ni < 2; ++ni) {
      b1fr[ni] = load_frag_b(sB1[cur], n_sub + 16 * ni, lane);
      b3fr[ni] = load_frag_b(sB3[cur], n_sub + 16 * ni, lane);
    }
#pragma unroll
    for (int mi = 0; mi < 2; ++mi)
#pragma unroll
      for (int ni = 0; ni < 2; ++ni) {
        acc1[mi][ni] = __builtin_amdgcn_wmma_f32_16x16x32_f16(
            false, afr[mi], false, b1fr[ni], (short)0, acc1[mi][ni], false, false);
        acc3[mi][ni] = __builtin_amdgcn_wmma_f32_16x16x32_f16(
            false, afr[mi], false, b3fr[ni], (short)0, acc3[mi][ni], false, false);
      }

    if (more) stage(cur ^ 1);
    cur ^= 1;
  }

  // epilogue: h = silu(a1) * a3 (fast rcp sigmoid), store f16
#pragma unroll
  for (int mi = 0; mi < 2; ++mi)
#pragma unroll
    for (int ni = 0; ni < 2; ++ni) {
#pragma unroll
      for (int r = 0; r < 8; ++r) {
        int grow = m0 + m_sub + 16 * mi + r + ((lane >> 4) & 1) * 8;
        int n = n0 + n_sub + 16 * ni + (lane & 15);
        float a1 = acc1[mi][ni][r];
        float a3 = acc3[mi][ni][r];
        float h = a1 * __builtin_amdgcn_rcpf(1.f + __expf(-a1)) * a3;
        H[(size_t)grow * IDIM + n] = (_Float16)h;
      }
    }
}

// -------------------- GEMM 2: partial[token][slot] = w_gate * (H * W2^T) --------------------
__global__ __launch_bounds__(128)
void moe_ffn2(const _Float16* __restrict__ H, const float* __restrict__ w2,
              const int* __restrict__ aoff, const int* __restrict__ counts,
              const int* __restrict__ row_token, const float* __restrict__ row_w,
              const int* __restrict__ row_slot, float* __restrict__ partial) {
  __shared__ __align__(16) _Float16 sA[2][64 * 32];
  __shared__ __align__(16) _Float16 sB[2][64 * 32];

  const int m0 = blockIdx.y * 64;   // padded-row tile
  const int n0 = blockIdx.x * 64;   // D-channel tile
  if (m0 >= aoff[8]) return;
  int e = 0;
#pragma unroll
  for (int k = 1; k < NEXP; ++k) if (m0 >= aoff[k]) e = k;
  if (m0 - aoff[e] >= counts[e]) return;

  const int tid = threadIdx.x;
  const int lane = tid & 31;
  const int wave = tid >> 5;
  const int m_sub = (wave & 1) * 32;
  const int n_sub = (wave >> 1) * 32;
  const float* w2e = w2 + (size_t)e * DDIM * IDIM;

  // A tile (f16, no conversion needed): async copy global->LDS, ASYNCcnt-tracked.
  // 64 rows x 32 halves = 256 16B chunks / 128 threads = 2 per thread.
  int rowA[2], colA[2];
#pragma unroll
  for (int it = 0; it < 2; ++it) {
    int task = tid + it * 128;
    rowA[it] = task >> 2;
    colA[it] = (task & 3) << 3;
  }
  auto async_copy_a = [&](int b, int k0) {
#pragma unroll
    for (int it = 0; it < 2; ++it) {
      unsigned int ldsd = (unsigned int)(size_t)&sA[b][rowA[it] * 32 + colA[it]];
      const _Float16* src = H + (size_t)(m0 + rowA[it]) * IDIM + k0 + colA[it];
      asm volatile("global_load_async_to_lds_b128 %0, %1, off"
                   :: "v"(ldsd), "v"(src) : "memory");
    }
  };

  // B tile (f32 -> f16 via VGPRs)
  int rowB[4], colB[4];
#pragma unroll
  for (int it = 0; it < 4; ++it) {
    int task = tid + it * 128;
    rowB[it] = task >> 3;
    colB[it] = (task & 7) << 2;
  }
  float4 pb[4];
  auto prefetch_b = [&](int k0) {
#pragma unroll
    for (int it = 0; it < 4; ++it)
      pb[it] = *(const float4*)(w2e + (size_t)(n0 + rowB[it]) * IDIM + k0 + colB[it]);
  };
  auto stage_b = [&](int b) {
#pragma unroll
    for (int it = 0; it < 4; ++it) {
      int o = rowB[it] * 32 + colB[it];
      sB[b][o + 0] = (_Float16)pb[it].x; sB[b][o + 1] = (_Float16)pb[it].y;
      sB[b][o + 2] = (_Float16)pb[it].z; sB[b][o + 3] = (_Float16)pb[it].w;
    }
  };

  const v8f vzero = {0.f, 0.f, 0.f, 0.f, 0.f, 0.f, 0.f, 0.f};
  v8f acc[2][2];
#pragma unroll
  for (int mi = 0; mi < 2; ++mi)
#pragma unroll
    for (int ni = 0; ni < 2; ++ni) acc[mi][ni] = vzero;

  async_copy_a(0, 0);
  prefetch_b(0);
  stage_b(0);
  asm volatile("s_wait_asynccnt 0x0" ::: "memory");
  int cur = 0;

  for (int k0 = 0; k0 < IDIM; k0 += 32) {
    __syncthreads();                       // publish buf[cur]
    const bool more = (k0 + 32) < IDIM;
    if (more) {
      async_copy_a(cur ^ 1, k0 + 32);      // async A tile for next step
      prefetch_b(k0 + 32);
    }

    v16h afr[2], bfr[2];
#pragma unroll
    for (int mi = 0; mi < 2; ++mi) afr[mi] = load_frag_a(sA[cur], m_sub + 16 * mi, lane);
#pragma unroll
    for (int ni = 0; ni < 2; ++ni) bfr[ni] = load_frag_b(sB[cur], n_sub + 16 * ni, lane);
#pragma unroll
    for (int mi = 0; mi < 2; ++mi)
#pragma unroll
      for (int ni = 0; ni < 2; ++ni)
        acc[mi][ni] = __builtin_amdgcn_wmma_f32_16x16x32_f16(
            false, afr[mi], false, bfr[ni], (short)0, acc[mi][ni], false, false);

    if (more) stage_b(cur ^ 1);
    asm volatile("s_wait_asynccnt 0x0" ::: "memory");  // A[next] landed before next barrier
    cur ^= 1;
  }

#pragma unroll
  for (int mi = 0; mi < 2; ++mi)
#pragma unroll
    for (int ni = 0; ni < 2; ++ni) {
#pragma unroll
      for (int r = 0; r < 8; ++r) {
        int grow = m0 + m_sub + 16 * mi + r + ((lane >> 4) & 1) * 8;
        int tok = row_token[grow];
        if (tok >= 0) {
          int n = n0 + n_sub + 16 * ni + (lane & 15);
          float v = acc[mi][ni][r] * row_w[grow];
          partial[((size_t)tok * 2 + row_slot[grow]) * DDIM + n] = v;
        }
      }
    }
}

// -------------------- deterministic 2-slot combine --------------------
__global__ void moe_combine(const float* __restrict__ p, float* __restrict__ out) {
  int i = blockIdx.x * blockDim.x + threadIdx.x;
  const int total = T_TOK * DDIM;
  for (; i < total; i += gridDim.x * blockDim.x) {
    int t = i / DDIM, d = i - t * DDIM;
    out[i] = p[(size_t)(t * 2) * DDIM + d] + p[(size_t)(t * 2 + 1) * DDIM + d];
  }
}

extern "C" void kernel_launch(void* const* d_in, const int* in_sizes, int n_in,
                              void* d_out, int out_size, void* d_ws, size_t ws_size,
                              hipStream_t stream) {
  const float* x  = (const float*)d_in[0];
  const float* gw = (const float*)d_in[1];
  const float* w1 = (const float*)d_in[2];
  const float* w2 = (const float*)d_in[3];
  const float* w3 = (const float*)d_in[4];
  float* out = (float*)d_out;
  char* ws = (char*)d_ws;

  int*      counts   = (int*)(ws + WS_COUNTS);
  int*      cursors  = (int*)(ws + WS_CURSORS);
  int*      aoff     = (int*)(ws + WS_AOFF);
  int*      tok_e    = (int*)(ws + WS_TOKE);
  float*    tok_w    = (float*)(ws + WS_TOKW);
  int*      row_tok  = (int*)(ws + WS_ROWTOK);
  float*    row_w    = (float*)(ws + WS_ROWW);
  int*      row_slot = (int*)(ws + WS_ROWSLOT);
  _Float16* Hbuf     = (_Float16*)(ws + WS_H);
  float*    partial  = (float*)(ws + WS_PARTIAL);

  moe_init<<<(RMAX + 255) / 256, 256, 0, stream>>>(counts, row_tok);
  moe_gate<<<T_TOK / 8, 256, 0, stream>>>(x, gw, counts, tok_e, tok_w);
  moe_scan<<<1, 1, 0, stream>>>(counts, aoff);
  moe_scatter<<<T_TOK / 256, 256, 0, stream>>>(tok_e, tok_w, aoff, cursors,
                                               row_tok, row_w, row_slot);
  dim3 g1(IDIM / 64, RMAX / 64);   // 64 x 72
  moe_ffn1<<<g1, 128, 0, stream>>>(x, w1, w3, aoff, counts, row_tok, Hbuf);
  dim3 g2(DDIM / 64, RMAX / 64);   // 32 x 72
  moe_ffn2<<<g2, 128, 0, stream>>>(Hbuf, w2, aoff, counts, row_tok, row_w,
                                   row_slot, partial);
  moe_combine<<<2048, 256, 0, stream>>>(partial, out);
  (void)in_sizes; (void)n_in; (void)out_size; (void)ws_size;
}
